// HS_layer_27178553049381
// MI455X (gfx1250) — matrix operand order repaired and checked by reading, not compiled, not run on hardware
//
#include <hip/hip_runtime.h>
#include <hip/hip_bf16.h>

typedef _Float16 v8h  __attribute__((ext_vector_type(8)));
typedef _Float16 v16h __attribute__((ext_vector_type(16)));
typedef float    v8f  __attribute__((ext_vector_type(8)));
typedef unsigned int u32x4 __attribute__((ext_vector_type(4)));
typedef int          i32x4 __attribute__((ext_vector_type(4)));
typedef int          i32x8 __attribute__((ext_vector_type(8)));

#define BS 8
#define NPTS 2048
#define CH 128
#define KNN 16
#define FMW 512      // (SUPPORT+1)*OUT_CH
#define LPAD 136     // 128 + 8 halves row pad: keeps 16B alignment, kills bank conflicts

// ---------------------------------------------------------------------------
// Tensor Data Mover staging: one 2D descriptor per panel.  data_size=4B,
// rows of 64 dwords (256B), LDS pad of 4 dwords per 64 dwords -> LPAD layout.
// ---------------------------------------------------------------------------
#if __has_builtin(__builtin_amdgcn_tensor_load_to_lds)
#define USE_TDM 1

__device__ __forceinline__ unsigned lds_offset_of(const void* p) {
    return (unsigned)(size_t)(__attribute__((address_space(3))) const void*)p;
}

__device__ __forceinline__ void tdm_load_panel(const void* gptr, unsigned lds_off, int rows) {
    unsigned long long ga = (unsigned long long)(size_t)gptr;
    const int row_dw = 64;            // 128 halves = 64 dwords per row
    u32x4 g0;
    g0[0] = 1u;                                            // count=1, user descriptor
    g0[1] = lds_off;                                       // lds_addr (bytes)
    g0[2] = (unsigned)(ga & 0xFFFFFFFFu);                  // global_addr lo
    g0[3] = (unsigned)((ga >> 32) & 0x01FFFFFFu) | (2u << 30); // global_addr hi | type=2
    i32x8 g1;
    g1[0] = (2 << 16)                                      // data_size = 4B
          | (1 << 20)                                      // pad_enable
          | (5 << 22)                                      // pad_interval: 64 dwords
          | (3 << 25);                                     // pad_amount: 4 dwords
    g1[1] = (row_dw & 0xFFFF) << 16;                       // tensor_dim0 lo16 (abar addr=0)
    g1[2] = ((rows & 0xFFFF) << 16);                       // tensor_dim0 hi=0 | tensor_dim1 lo16
    g1[3] = ((row_dw & 0xFFFF) << 16);                     // tensor_dim1 hi=0 | tile_dim0
    g1[4] = (rows & 0xFFFF);                               // tile_dim1 | tile_dim2=0
    g1[5] = row_dw;                                        // tensor_dim0_stride lo32
    g1[6] = 0;
    g1[7] = 0;
    i32x4 z4 = {0, 0, 0, 0};
#if __clang_major__ >= 23
    i32x8 z8 = {0, 0, 0, 0, 0, 0, 0, 0};
    __builtin_amdgcn_tensor_load_to_lds(g0, g1, z4, z4, z8, 0);
#else
    __builtin_amdgcn_tensor_load_to_lds(g0, g1, z4, z4, 0);
#endif
}
#else
#define USE_TDM 0
#endif

// ---------------------------------------------------------------------------
// WMMA fragment loads from LDS (16x16x32 f16), per ISA 7.12.2 layouts:
// A (16x32 MxK): lanes 0-15 -> M=lane,    K = {k0+0..7, k0+16..23}
//                lanes16-31 -> M=lane-16, K = {k0+8..15, k0+24..31}
// B (32x16 KxN): lanes 0-15 -> N=lane,    K = {k0+0..7, k0+16..23}
//                lanes16-31 -> N=lane-16, K = {k0+8..15, k0+24..31}
// Both panels are stored row-major (rows x LPAD halves) with K contiguous, so
// this computes C = A_rows * B_rows^T (dot of row i with row j).
// ---------------------------------------------------------------------------
__device__ __forceinline__ v16h lds_a_frag(const _Float16* sA, int row0, int k0, int lane) {
    int r  = lane & 15;
    int kh = (lane >> 4) & 1;
    const _Float16* p = sA + (row0 + r) * LPAD + k0 + kh * 8;
    v8h lo = *(const v8h*)(p);
    v8h hi = *(const v8h*)(p + 16);
    v16h out;
#pragma unroll
    for (int i = 0; i < 8; ++i) { out[i] = lo[i]; out[i + 8] = hi[i]; }
    return out;
}

__device__ __forceinline__ v16h lds_b_frag(const _Float16* sB, int col0, int k0, int lane) {
    int c  = lane & 15;
    int kh = (lane >> 4) & 1;
    const _Float16* p = sB + (col0 + c) * LPAD + k0 + kh * 8;
    v8h lo = *(const v8h*)(p);
    v8h hi = *(const v8h*)(p + 16);
    v16h out;
#pragma unroll
    for (int i = 0; i < 8; ++i) { out[i] = lo[i]; out[i + 8] = hi[i]; }
    return out;
}

// ---------------------------------------------------------------------------
// Block-cooperative GEMM: C[MxN] = A(Mx128 f16) * B(Nx128 f16)^T (+bias[n]).
// Block = 256 threads (8 waves), output tile 128x64.  A-panel (128x128) and
// B-panel (64x128) staged in LDS (52 KB) by the Tensor Data Mover (wave 0
// issues two descriptors, waits TENSORcnt, then workgroup barrier), each wave
// owns one 16-row strip and sweeps 4 j-tiles -> 16 v_wmma per wave.
// All dims multiples of the tile.  K fixed at 128.
// ---------------------------------------------------------------------------
__global__ void __launch_bounds__(256)
gemm_blk_f16(const _Float16* __restrict__ A,
             const _Float16* __restrict__ B,
             const float* __restrict__ bias,
             float* __restrict__ C, int ldc) {
    __shared__ _Float16 sA[128 * LPAD];
    __shared__ _Float16 sB[64 * LPAD];
    const int tid = threadIdx.x;
    const int i0 = blockIdx.x * 128;
    const int j0 = blockIdx.y * 64;
    const _Float16* Ag = A + (size_t)i0 * 128;
    const _Float16* Bg = B + (size_t)j0 * 128;

#if USE_TDM
    if (tid < 32) {
        tdm_load_panel(Ag, lds_offset_of(sA), 128);
        tdm_load_panel(Bg, lds_offset_of(sB), 64);
        __builtin_amdgcn_s_wait_tensorcnt((short)0);
    }
    __syncthreads();
#else
#pragma unroll 2
    for (int t = tid; t < 128 * 16; t += 256) {
        int row = t >> 4, seg = t & 15;
        v8h v = *(const v8h*)(Ag + (size_t)row * 128 + seg * 8);
        *(v8h*)(sA + row * LPAD + seg * 8) = v;
    }
#pragma unroll 2
    for (int t = tid; t < 64 * 16; t += 256) {
        int row = t >> 4, seg = t & 15;
        v8h v = *(const v8h*)(Bg + (size_t)row * 128 + seg * 8);
        *(v8h*)(sB + row * LPAD + seg * 8) = v;
    }
    __syncthreads();
#endif

    const int wave = tid >> 5;
    const int lane = tid & 31;
    const int istrip = wave * 16;

    v8f acc0 = {}, acc1 = {}, acc2 = {}, acc3 = {};
#pragma unroll
    for (int kc = 0; kc < 128; kc += 32) {
        v16h a  = lds_a_frag(sA, istrip, kc, lane);
        v16h b0 = lds_b_frag(sB,  0, kc, lane);
        v16h b1 = lds_b_frag(sB, 16, kc, lane);
        v16h b2 = lds_b_frag(sB, 32, kc, lane);
        v16h b3 = lds_b_frag(sB, 48, kc, lane);
        acc0 = __builtin_amdgcn_wmma_f32_16x16x32_f16(false, a, false, b0, (short)0, acc0, false, false);
        acc1 = __builtin_amdgcn_wmma_f32_16x16x32_f16(false, a, false, b1, (short)0, acc1, false, false);
        acc2 = __builtin_amdgcn_wmma_f32_16x16x32_f16(false, a, false, b2, (short)0, acc2, false, false);
        acc3 = __builtin_amdgcn_wmma_f32_16x16x32_f16(false, a, false, b3, (short)0, acc3, false, false);
    }

    const int mb = i0 + istrip + ((lane >> 4) & 1) * 8;
    const int nb = lane & 15;
    v8f accs[4] = {acc0, acc1, acc2, acc3};
#pragma unroll
    for (int jt = 0; jt < 4; ++jt) {
        int n = j0 + jt * 16 + nb;
        float bv = bias ? bias[n] : 0.0f;
#pragma unroll
        for (int r = 0; r < 8; ++r) {
            C[(size_t)(mb + r) * ldc + n] = accs[jt][r] + bv;
        }
    }
}

// ---------------------------------------------------------------------------
// Prep kernels
// ---------------------------------------------------------------------------
__global__ void k_convert_feat(const float* __restrict__ f, _Float16* __restrict__ f16, int total) {
    int i = blockIdx.x * blockDim.x + threadIdx.x;
    if (i < total) f16[i] = (_Float16)f[i];
}

__global__ void k_row_sq(const float* __restrict__ f, float* __restrict__ sq) {
    int i = blockIdx.x * blockDim.x + threadIdx.x;
    if (i >= BS * NPTS) return;
    const float* row = f + (size_t)i * CH;
    float s = 0.0f;
#pragma unroll 8
    for (int c = 0; c < CH; ++c) { float v = row[c]; s += v * v; }
    sq[i] = s;
}

__global__ void k_prep_small(const float* __restrict__ weights,   // (128,512)
                             const float* __restrict__ ste_w,     // (128,128)
                             const float* __restrict__ conv2_w,   // (128,256)
                             const float* __restrict__ directions,// (3,384)
                             _Float16* __restrict__ wT16,         // (512,128)
                             _Float16* __restrict__ stew16,       // (128,128)
                             _Float16* __restrict__ conv2a16,     // (128,128)
                             float* __restrict__ sdn) {           // (3,384)
    int i = blockIdx.x * blockDim.x + threadIdx.x;
    if (i < FMW * CH) {
        int m = i / CH, k = i % CH;
        wT16[i] = (_Float16)weights[(size_t)k * FMW + m];
    }
    if (i < CH * CH) {
        stew16[i] = (_Float16)ste_w[i];
        int o = i / CH, c = i % CH;
        conv2a16[i] = (_Float16)conv2_w[(size_t)o * 256 + c];
    }
    if (i < 384) {
        float a = directions[i], b = directions[384 + i], c = directions[768 + i];
        float inv = 1.0f / fmaxf(sqrtf(a * a + b * b + c * c), 1e-12f);
        sdn[i] = a * inv; sdn[384 + i] = b * inv; sdn[768 + i] = c * inv;
    }
}

// ---------------------------------------------------------------------------
// Top-17 selection (feature space) from the Gram matrix of one batch.
// dist(i,j) = sq_i + sq_j - 2*G[i][j]; keep 17 smallest, drop first (self).
// ---------------------------------------------------------------------------
__global__ void k_topk_feat(const float* __restrict__ G, const float* __restrict__ sq,
                            int b, int* __restrict__ idx_out) {
    int i = blockIdx.x * blockDim.x + threadIdx.x;
    if (i >= NPTS) return;
    const float* g   = G + (size_t)i * NPTS;
    const float* sqb = sq + (size_t)b * NPTS;
    float sqi = sqb[i];
    float d[17]; int id[17];
#pragma unroll
    for (int t = 0; t < 17; ++t) { d[t] = 3.0e38f; id[t] = 0; }
    float worst = 3.0e38f;
    for (int j = 0; j < NPTS; ++j) {
        float dist = sqi + sqb[j] - 2.0f * g[j];
        if (dist < worst) {
            int p = 16;
            while (p > 0 && d[p - 1] > dist) { d[p] = d[p - 1]; id[p] = id[p - 1]; --p; }
            d[p] = dist; id[p] = j;
            worst = d[16];
        }
    }
    int* out = idx_out + ((size_t)b * NPTS + i) * KNN;
    for (int t = 0; t < KNN; ++t) out[t] = id[t + 1];
}

// Vertex-space kNN (K=3): direct squared distance, same top-17 scheme.
__global__ void k_topk_vert(const float* __restrict__ vertices, int* __restrict__ idx_out) {
    int i = blockIdx.x * blockDim.x + threadIdx.x;
    int b = blockIdx.y;
    if (i >= NPTS) return;
    const float* vb = vertices + (size_t)b * NPTS * 3;
    float x = vb[i * 3 + 0], y = vb[i * 3 + 1], z = vb[i * 3 + 2];
    float d[17]; int id[17];
#pragma unroll
    for (int t = 0; t < 17; ++t) { d[t] = 3.0e38f; id[t] = 0; }
    float worst = 3.0e38f;
    for (int j = 0; j < NPTS; ++j) {
        float dx = vb[j * 3 + 0] - x, dy = vb[j * 3 + 1] - y, dz = vb[j * 3 + 2] - z;
        float dist = dx * dx + dy * dy + dz * dz;
        if (dist < worst) {
            int p = 16;
            while (p > 0 && d[p - 1] > dist) { d[p] = d[p - 1]; id[p] = id[p - 1]; --p; }
            d[p] = dist; id[p] = j;
            worst = d[16];
        }
    }
    int* out = idx_out + ((size_t)b * NPTS + i) * KNN;
    for (int t = 0; t < KNN; ++t) out[t] = id[t + 1];
}

// ---------------------------------------------------------------------------
// Aggregation: theta = relu(rf_norm . sdn), act = mean_s max_k (theta*support),
// feature = fm_center + act.   One block (128 threads) per (b,n).
// ---------------------------------------------------------------------------
__global__ void k_aggregate(const float* __restrict__ vertices,
                            const float* __restrict__ fm,      // (BS*N,512)
                            const int*   __restrict__ idx1,
                            const float* __restrict__ sdn,     // (3,384)
                            float* __restrict__ feature,
                            _Float16* __restrict__ feature16) {
    int n = blockIdx.x, b = blockIdx.y, o = threadIdx.x;
    __shared__ int   sidx[KNN];
    __shared__ float srf[KNN][3];
    const float* vb = vertices + (size_t)b * NPTS * 3;
    if (o < KNN) {
        int nb = idx1[((size_t)b * NPTS + n) * KNN + o];
        sidx[o] = nb;
        float dx = vb[nb * 3 + 0] - vb[n * 3 + 0];
        float dy = vb[nb * 3 + 1] - vb[n * 3 + 1];
        float dz = vb[nb * 3 + 2] - vb[n * 3 + 2];
        float inv = 1.0f / fmaxf(sqrtf(dx * dx + dy * dy + dz * dz), 1e-12f);
        srf[o][0] = dx * inv; srf[o][1] = dy * inv; srf[o][2] = dz * inv;
    }
    __syncthreads();
    float s00 = sdn[0 * 384 + o],        s10 = sdn[1 * 384 + o],        s20 = sdn[2 * 384 + o];
    float s01 = sdn[0 * 384 + 128 + o],  s11 = sdn[1 * 384 + 128 + o],  s21 = sdn[2 * 384 + 128 + o];
    float s02 = sdn[0 * 384 + 256 + o],  s12 = sdn[1 * 384 + 256 + o],  s22 = sdn[2 * 384 + 256 + o];
    float mx0 = -3.0e38f, mx1 = -3.0e38f, mx2 = -3.0e38f;
    const float* fmb = fm + (size_t)b * NPTS * FMW;
#pragma unroll 4
    for (int kk = 0; kk < KNN; ++kk) {
        float r0 = srf[kk][0], r1 = srf[kk][1], r2 = srf[kk][2];
        const float* fr = fmb + (size_t)sidx[kk] * FMW + CH;
        float t0 = fmaxf(r0 * s00 + r1 * s10 + r2 * s20, 0.0f);
        float t1 = fmaxf(r0 * s01 + r1 * s11 + r2 * s21, 0.0f);
        float t2 = fmaxf(r0 * s02 + r1 * s12 + r2 * s22, 0.0f);
        mx0 = fmaxf(mx0, t0 * fr[o]);
        mx1 = fmaxf(mx1, t1 * fr[CH + o]);
        mx2 = fmaxf(mx2, t2 * fr[2 * CH + o]);
    }
    float val = fmb[(size_t)n * FMW + o] + (mx0 + mx1 + mx2) * (1.0f / 3.0f);
    size_t oidx = ((size_t)b * NPTS + n) * CH + o;
    feature[oidx]   = val;
    feature16[oidx] = (_Float16)val;
}

// pooled[b,n,o] = max_k feature[b, idx2[b,n,k], o]
__global__ void k_pool_max(const float* __restrict__ feature, const int* __restrict__ idx2,
                           float* __restrict__ pooled) {
    int n = blockIdx.x, b = blockIdx.y, o = threadIdx.x;
    __shared__ int sidx[KNN];
    if (o < KNN) sidx[o] = idx2[((size_t)b * NPTS + n) * KNN + o];
    __syncthreads();
    float mx = -3.0e38f;
#pragma unroll
    for (int kk = 0; kk < KNN; ++kk)
        mx = fmaxf(mx, feature[((size_t)b * NPTS + sidx[kk]) * CH + o]);
    pooled[((size_t)b * NPTS + n) * CH + o] = mx;
}

// f_global[b] = mean_n pooled[b];  g2[b,o] = sum_j f_global[b,j]*conv2_w[o,128+j]
__global__ void k_global_g2(const float* __restrict__ pooled, const float* __restrict__ conv2_w,
                            float* __restrict__ g2) {
    int b = blockIdx.x, o = threadIdx.x;
    __shared__ float fg[CH];
    const float* p = pooled + (size_t)b * NPTS * CH + o;
    float s = 0.0f;
    for (int n = 0; n < NPTS; ++n) s += p[(size_t)n * CH];
    fg[o] = s * (1.0f / (float)NPTS);
    __syncthreads();
    float acc = 0.0f;
#pragma unroll 8
    for (int j = 0; j < CH; ++j) acc += fg[j] * conv2_w[(size_t)o * 256 + CH + j];
    g2[b * CH + o] = acc;
}

// out = fuse(conv2 part A) + g2 (conv2 part B) + feature + f_ste
__global__ void k_final_add(const float* __restrict__ fuse, const float* __restrict__ feature,
                            const float* __restrict__ f_ste, const float* __restrict__ g2,
                            float* __restrict__ out, int total) {
    int i = blockIdx.x * blockDim.x + threadIdx.x;
    if (i >= total) return;
    int o = i & (CH - 1);
    int b = i >> 18;  // / (NPTS*CH) == / 262144
    out[i] = fuse[i] + feature[i] + f_ste[i] + g2[b * CH + o];
}

// ---------------------------------------------------------------------------
extern "C" void kernel_launch(void* const* d_in, const int* in_sizes, int n_in,
                              void* d_out, int out_size, void* d_ws, size_t ws_size,
                              hipStream_t stream) {
    (void)in_sizes; (void)n_in; (void)out_size; (void)ws_size;
    const float* vertices   = (const float*)d_in[0];
    const float* feature_map= (const float*)d_in[1];
    const float* weights    = (const float*)d_in[2];
    const float* bias       = (const float*)d_in[3];
    const float* directions = (const float*)d_in[4];
    const float* ste_w      = (const float*)d_in[5];
    const float* conv2_w    = (const float*)d_in[6];
    float* out = (float*)d_out;

    char* ws = (char*)d_ws;
    size_t off = 0;
    auto alloc = [&](size_t bytes) -> void* {
        void* p = ws + off;
        off = (off + bytes + 255) & ~(size_t)255;
        return p;
    };
    const int ROWS = BS * NPTS;                       // 16384
    _Float16* feat16    = (_Float16*)alloc((size_t)ROWS * CH * 2);
    _Float16* wT16      = (_Float16*)alloc((size_t)FMW * CH * 2);
    _Float16* stew16    = (_Float16*)alloc((size_t)CH * CH * 2);
    _Float16* conv2a16  = (_Float16*)alloc((size_t)CH * CH * 2);
    float*    sdn       = (float*)   alloc((size_t)3 * 384 * 4);
    float*    sq        = (float*)   alloc((size_t)ROWS * 4);
    float*    G         = (float*)   alloc((size_t)NPTS * NPTS * 4);
    int*      idx1      = (int*)     alloc((size_t)ROWS * KNN * 4);
    int*      idx2      = (int*)     alloc((size_t)ROWS * KNN * 4);
    float*    fm        = (float*)   alloc((size_t)ROWS * FMW * 4);
    float*    f_ste     = (float*)   alloc((size_t)ROWS * CH * 4);
    float*    feature   = (float*)   alloc((size_t)ROWS * CH * 4);
    _Float16* feature16 = (_Float16*)alloc((size_t)ROWS * CH * 2);
    float*    pooled    = (float*)   alloc((size_t)ROWS * CH * 4);
    float*    g2        = (float*)   alloc((size_t)BS * CH * 4);
    float*    fuse      = (float*)   alloc((size_t)ROWS * CH * 4);

    const int total_fc = ROWS * CH;                   // 2M

    // --- prep ---
    k_convert_feat<<<(total_fc + 255) / 256, 256, 0, stream>>>(feature_map, feat16, total_fc);
    k_row_sq<<<(ROWS + 255) / 256, 256, 0, stream>>>(feature_map, sq);
    k_prep_small<<<(FMW * CH + 255) / 256, 256, 0, stream>>>(weights, ste_w, conv2_w, directions,
                                                             wT16, stew16, conv2a16, sdn);

    // --- feature-space kNN: per batch Gram matrix (WMMA, TDM-staged) + top-17 ---
    for (int b = 0; b < BS; ++b) {
        const _Float16* fb = feat16 + (size_t)b * NPTS * CH;
        gemm_blk_f16<<<dim3(NPTS / 128, NPTS / 64), 256, 0, stream>>>(fb, fb, nullptr, G, NPTS);
        k_topk_feat<<<(NPTS + 255) / 256, 256, 0, stream>>>(G, sq, b, idx1);
    }

    // --- vertex-space kNN ---
    k_topk_vert<<<dim3((NPTS + 255) / 256, BS), 256, 0, stream>>>(vertices, idx2);

    // --- fm = feature_map @ weights + bias (16384x512, K=128, WMMA) ---
    gemm_blk_f16<<<dim3(ROWS / 128, FMW / 64), 256, 0, stream>>>(feat16, wT16, bias, fm, FMW);

    // --- f_ste = feature_map @ ste_w^T (16384x128, WMMA) ---
    gemm_blk_f16<<<dim3(ROWS / 128, CH / 64), 256, 0, stream>>>(feat16, stew16, nullptr, f_ste, CH);

    // --- aggregation -> feature (+f16 copy) ---
    k_aggregate<<<dim3(NPTS, BS), CH, 0, stream>>>(vertices, fm, idx1, sdn, feature, feature16);

    // --- neighbor max pool + global mean + conv2 second half ---
    k_pool_max<<<dim3(NPTS, BS), CH, 0, stream>>>(feature, idx2, pooled);
    k_global_g2<<<BS, CH, 0, stream>>>(pooled, conv2_w, g2);

    // --- conv2 first half on feature (16384x128, K=128, WMMA) ---
    gemm_blk_f16<<<dim3(ROWS / 128, CH / 64), 256, 0, stream>>>(feature16, conv2a16, nullptr, fuse, CH);

    // --- final: fuse + g2 + feature + f_ste ---
    k_final_add<<<(total_fc + 255) / 256, 256, 0, stream>>>(fuse, feature, f_ste, g2, out, total_fc);
}
